// ConvTemporalGraphical_73134703116407
// MI455X (gfx1250) — matrix operand order repaired and checked by reading, not compiled, not run on hardware
//
#include <hip/hip_runtime.h>

// ---------------------------------------------------------------------------
// ST-GCN adaptive graph conv block on MI455X (gfx1250), bf16 WMMA pipeline.
//   K0 : pack weights / adjacency / fused bias, zero pads
//   KX : xb = bf16 transpose of x (channel-contiguous)
//   K1 : [a1|d1|conv] = Wcat[80,64] @ unfold(x)         (wmma bf16)
//   K2 : logits = (1/75) a1^T a1, K=8192                (wmma bf16)
//   K3 : column softmax -> attT (transposed, padded)
//   K4 : s = d1@att + sum_k conv_k@Aeff_k               (wmma bf16, 12/tile)
//   K5 : y = out_w @ s + BN + residual + relu           (wmma bf16)
// Fragment gathers are unconditional dword loads (merged by the compiler into
// b128 clauses); only the final K-block of a GEMM uses clamped addresses with
// value-selects to zero, so no branch ladders and no NaN leakage from pads.
// ---------------------------------------------------------------------------

typedef __bf16 bf16;
typedef __attribute__((ext_vector_type(2)))  __bf16 bf16x2;
typedef __attribute__((ext_vector_type(16))) __bf16 v16bf;
typedef __attribute__((ext_vector_type(2)))  float  f32x2;
typedef __attribute__((ext_vector_type(8)))  float  v8f;

#define WMMA_BF16(a, b, c) \
  __builtin_amdgcn_wmma_f32_16x16x32_bf16(false, (a), false, (b), (short)0, (c), false, false)

#define NB    32
#define C_IN  64
#define T_DIM 512
#define V_DIM 25
#define W3    75
#define C4    16
#define CO    64
#define PW    76                 // padded w stride of P2 rows
#define AKP   96                 // padded K (w' dim) stride for attT / aeffT
#define KCT   (C4 * T_DIM)       // 8192, Gram K
#define TV    (T_DIM * V_DIM)    // 12800

// A-fragment K offset for element pair i (16x32 bf16 A, wave32)
__device__ __forceinline__ int ak_of(int i, int hi) {
  return (i < 4) ? (8 * hi + 2 * i) : (16 + 8 * hi + 2 * (i - 4));
}
// B-fragment K offset for element pair i (32x16 bf16 B, wave32)
__device__ __forceinline__ int bk_of(int i, int hi) { return (i < 4) ? (8 * hi + 2 * i) : (16 + 8 * hi + 2 * (i - 4)); }

__device__ __forceinline__ bf16x2 ld2(const bf16* p) { return *(const bf16x2*)p; }

// ---------------------------------------------------------------------------
// K0: pack weights/adjacency (transposed+padded), fused bias, zero attT
// ---------------------------------------------------------------------------
__global__ void k_prep(const float* __restrict__ A, const float* __restrict__ PA,
                       const float* __restrict__ wa, const float* __restrict__ wd,
                       const float* __restrict__ wc, const float* __restrict__ ow,
                       const float* __restrict__ ab, const float* __restrict__ db,
                       const float* __restrict__ cb,
                       bf16* __restrict__ aeffT, bf16* __restrict__ wcat,
                       bf16* __restrict__ w2, bf16* __restrict__ attT,
                       float* __restrict__ bcat) {
  int idx = blockIdx.x * blockDim.x + threadIdx.x;
  int stride = gridDim.x * blockDim.x;
  // aeffT[kk][w][w'] = Aeff[kk][w'][w], zero-padded to 80x96
  for (int i = idx; i < 3 * 80 * AKP; i += stride) {
    int kk = i / (80 * AKP), w = (i / AKP) % 80, k = i % AKP;
    float v = 0.f;
    if (w < W3 && k < W3) { int a = (kk * W3 + k) * W3 + w; v = A[a] + PA[a]; }
    aeffT[i] = (bf16)v;
  }
  for (int i = idx; i < 80 * C_IN; i += stride) {
    int o = i / C_IN, c = i % C_IN;
    float v;
    if (o < 16)      v = wa[o * C_IN + c];
    else if (o < 32) v = wd[(o - 16) * C_IN + c];
    else             v = wc[(o - 32) * C_IN + c];
    wcat[i] = (bf16)v;
  }
  for (int i = idx; i < 80; i += stride)
    bcat[i] = (i < 16) ? ab[i] : (i < 32) ? db[i - 16] : cb[i - 32];
  for (int i = idx; i < CO * 64; i += stride) {
    int o = i / 64, k = i % 64;
    w2[i] = (bf16)((k < 48) ? ow[o * 48 + k] : 0.f);
  }
  for (int i = idx; i < NB * 80 * AKP; i += stride) attT[i] = (bf16)0.f;
}

// ---------------------------------------------------------------------------
// KX: xb[n][t*V+v][c] = bf16(x[n][c][t][v])  (channel-contiguous transpose)
// ---------------------------------------------------------------------------
__global__ void k_xpose(const float* __restrict__ x, bf16* __restrict__ xb) {
  long i = (long)blockIdx.x * blockDim.x + threadIdx.x;
  if (i >= (long)NB * TV * C_IN) return;
  int c = (int)(i & 63);
  long p = i >> 6;
  int v = (int)(p % V_DIM);
  int t = (int)((p / V_DIM) % T_DIM);
  int n = (int)(p / TV);
  xb[i] = (bf16)x[(((long)n * C_IN + c) * T_DIM + t) * V_DIM + v];
}

// ---------------------------------------------------------------------------
// K1: [a1|d1|conv] = Wcat @ unfold(x); a1 -> a1t (K-major), rest -> P2
// ---------------------------------------------------------------------------
__global__ void k_gemm1(const bf16* __restrict__ xb, const bf16* __restrict__ wcat,
                        const float* __restrict__ bcat, bf16* __restrict__ P2,
                        bf16* __restrict__ a1t) {
  const int lane = threadIdx.x & 31, lo = lane & 15, hi = lane >> 4;
  long tile = (long)blockIdx.x * (blockDim.x >> 5) + (threadIdx.x >> 5);
  if (tile >= (long)NB * 5 * 2400) return;
  int n = (int)(tile / (5 * 2400));
  int rem = (int)(tile % (5 * 2400));
  int rt = rem / 2400;  // row tile over the 80 output channels
  int ct = rem % 2400;  // col tile over T*W3

  int col = ct * 16 + lo;
  int t = col / W3, w = col % W3;
  int j = w / V_DIM, v = w % V_DIM;
  int tt = t + j - 1;                          // temporal unfold, pad=1
  bool colok = (tt >= 0) && (tt < T_DIM);
  int ttc = tt < 0 ? 0 : (tt >= T_DIM ? T_DIM - 1 : tt);
  const bf16* xrow = xb + (((long)n * TV) + (long)ttc * V_DIM + v) * C_IN;
  const bf16* arow = wcat + (rt * 16 + lo) * C_IN;

  v8f acc = {};
#pragma unroll
  for (int kb = 0; kb < C_IN; kb += 32) {
    v16bf a, b;
#pragma unroll
    for (int i = 0; i < 8; ++i) {
      bf16x2 ap = ld2(arow + kb + ak_of(i, hi));
      a[2 * i] = ap.x; a[2 * i + 1] = ap.y;
      unsigned u = *(const unsigned*)(xrow + kb + bk_of(i, hi));
      u = colok ? u : 0u;  // value select, not a guarded load
      bf16x2 bp = __builtin_bit_cast(bf16x2, u);
      b[2 * i] = bp.x; b[2 * i + 1] = bp.y;
    }
    acc = WMMA_BF16(a, b, acc);
  }
  if (rt == 0) {  // a1 rows, stored K-major for the Gram kernel (wave-uniform)
#pragma unroll
    for (int r = 0; r < 8; ++r) {
      int orow = r + 8 * hi;
      a1t[((long)n * 80 + w) * KCT + orow * T_DIM + t] = (bf16)(acc[r] + bcat[orow]);
    }
  } else {        // d1 rows 0..15, conv rows 16..63 of P2
#pragma unroll
    for (int r = 0; r < 8; ++r) {
      int orow = rt * 16 + r + 8 * hi;
      P2[(((long)n * 64 + (orow - 16)) * T_DIM + t) * PW + w] = (bf16)(acc[r] + bcat[orow]);
    }
  }
}

// ---------------------------------------------------------------------------
// K2: logits = (1/75) a1^T a1, K=8192; pure dword loads from a1t
// ---------------------------------------------------------------------------
__global__ void k_att(const bf16* __restrict__ a1t, float* __restrict__ logits) {
  const int lane = threadIdx.x & 31, lo = lane & 15, hi = lane >> 4;
  int tile = blockIdx.x * (blockDim.x >> 5) + (threadIdx.x >> 5);
  if (tile >= NB * 25) return;
  int n = tile / 25, rem = tile % 25;
  int ti = rem / 5, tj = rem % 5;
  int irow = ti * 16 + lo;
  int jcol = tj * 16 + lo;
  int irc = irow < W3 ? irow : W3 - 1;  // clamp; dup rows land in discarded D rows
  int jrc = jcol < W3 ? jcol : W3 - 1;
  const bf16* aRow = a1t + ((long)n * 80 + irc) * KCT;
  const bf16* bRow = a1t + ((long)n * 80 + jrc) * KCT;

  v8f acc = {};
  for (int kb = 0; kb < KCT; kb += 32) {
    __builtin_prefetch(aRow + kb + 256, 0, 1);
    __builtin_prefetch(bRow + kb + 256, 0, 1);
    v16bf a, b;
#pragma unroll
    for (int i = 0; i < 8; ++i) {
      bf16x2 ap = ld2(aRow + kb + ak_of(i, hi));
      a[2 * i] = ap.x; a[2 * i + 1] = ap.y;
      bf16x2 bp = ld2(bRow + kb + bk_of(i, hi));
      b[2 * i] = bp.x; b[2 * i + 1] = bp.y;
    }
    acc = WMMA_BF16(a, b, acc);
  }
  const float inv = 1.0f / (float)W3;
#pragma unroll
  for (int r = 0; r < 8; ++r) {
    int i = ti * 16 + r + 8 * hi;
    if (i < W3 && jcol < W3) logits[((long)n * W3 + i) * W3 + jcol] = acc[r] * inv;
  }
}

// ---------------------------------------------------------------------------
// K3: softmax over rows i per column j; write attT[n][j][i] (K-major, padded)
// ---------------------------------------------------------------------------
__global__ void k_softmax(const float* __restrict__ logits, bf16* __restrict__ attT) {
  int b = blockIdx.x;  // n*75 + j
  int n = b / W3, j = b % W3;
  int tid = threadIdx.x;  // 128 threads
  __shared__ float red[128];
  float v = (tid < W3) ? logits[((long)n * W3 + tid) * W3 + j] : -1e30f;
  red[tid] = v;
  __syncthreads();
  for (int s = 64; s > 0; s >>= 1) {
    if (tid < s) red[tid] = fmaxf(red[tid], red[tid + s]);
    __syncthreads();
  }
  float mx = red[0];
  __syncthreads();
  float e = (tid < W3) ? __expf(v - mx) : 0.f;
  red[tid] = e;
  __syncthreads();
  for (int s = 64; s > 0; s >>= 1) {
    if (tid < s) red[tid] += red[tid + s];
    __syncthreads();
  }
  float sum = red[0];
  if (tid < AKP)
    attT[((long)n * 80 + j) * AKP + tid] = (bf16)((tid < W3) ? e / sum : 0.f);
}

// ---------------------------------------------------------------------------
// K4: s = d1 @ att + sum_k conv_k @ Aeff_k  (4 accumulating GEMMs, K=96)
// ---------------------------------------------------------------------------
__global__ void k_gemm2(const bf16* __restrict__ P2, const bf16* __restrict__ attT,
                        const bf16* __restrict__ aeffT, float* __restrict__ s2) {
  const int lane = threadIdx.x & 31, lo = lane & 15, hi = lane >> 4;
  long tile = (long)blockIdx.x * (blockDim.x >> 5) + (threadIdx.x >> 5);
  if (tile >= (long)NB * 512 * 5) return;
  int n = (int)(tile / (512 * 5));
  int rem = (int)(tile % (512 * 5));
  int rt = rem / 5, jt = rem % 5;
  int c4 = rt >> 5;               // constant within tile (512 % 16 == 0)
  int tm = (rt * 16 + lo) & 511;  // A row's t index
  int col = jt * 16 + lo;         // output w column (<80)
  const bf16* Pn = P2 + (long)n * 64 * T_DIM * PW;
  const bf16* bAtt = attT + ((long)n * 80 + col) * AKP;  // B rows, K-major

  v8f acc = {};
#pragma unroll
  for (int src = 0; src < 4; ++src) {  // 0: d1@att, 1..3: conv_k@Aeff_k
    int prow = (src == 0) ? c4 : (16 + (src - 1) * 16 + c4);
    const bf16* Ar = Pn + ((long)prow * T_DIM + tm) * PW;
    const bf16* Br = (src == 0) ? bAtt : (aeffT + (((src - 1) * 80) + col) * (long)AKP);
#pragma unroll
    for (int kb = 0; kb < AKP; kb += 32) {
      v16bf a, b;
#pragma unroll
      for (int i = 0; i < 8; ++i) {
        int kA = kb + ak_of(i, hi);
        if (kb < 64) {  // compile-time: k <= 62 < 75, plain contiguous loads
          bf16x2 ap = ld2(Ar + kA);
          a[2 * i] = ap.x; a[2 * i + 1] = ap.y;
        } else {        // last block: clamp address, select value to zero
          int kc = kA < (W3 - 1) ? kA : (W3 - 1);
          bf16x2 ap = ld2(Ar + kc);
          a[2 * i]     = (kA < W3)     ? ap.x : (bf16)0.f;
          a[2 * i + 1] = (kA + 1 < W3) ? ap.y : (bf16)0.f;
        }
        bf16x2 bp = ld2(Br + kb + bk_of(i, hi));  // B rows >=75 are true zeros
        b[2 * i] = bp.x; b[2 * i + 1] = bp.y;
      }
      acc = WMMA_BF16(a, b, acc);
    }
  }
#pragma unroll
  for (int r = 0; r < 8; ++r) {
    int tq = (rt * 16 + r + 8 * hi) & 511;
    if (col < W3) {
      int jq = col / V_DIM, vq = col % V_DIM;  // k-major store for K5
      s2[(((long)n * T_DIM + tq) * V_DIM + vq) * 48 + c4 * 3 + jq] = acc[r];
    }
  }
}

// ---------------------------------------------------------------------------
// K5: y = out_w @ s (collapse window of 3) + BN + residual + relu
// ---------------------------------------------------------------------------
__global__ void k_out(const float* __restrict__ s2, const bf16* __restrict__ w2,
                      const float* __restrict__ outb, const float* __restrict__ gamma,
                      const float* __restrict__ beta, const float* __restrict__ mean,
                      const float* __restrict__ var, const float* __restrict__ x,
                      float* __restrict__ y) {
  const int lane = threadIdx.x & 31, lo = lane & 15, hi = lane >> 4;
  long tile = (long)blockIdx.x * (blockDim.x >> 5) + (threadIdx.x >> 5);
  if (tile >= (long)NB * 4 * 800) return;
  int n = (int)(tile / 3200);
  int rem = (int)(tile % 3200);
  int ot = rem / 800, ct = rem % 800;
  int col = ct * 16 + lo;
  int t = col / V_DIM, v = col % V_DIM;
  const float* srow = s2 + (((long)n * T_DIM + t) * V_DIM + v) * 48;
  const bf16* arow = w2 + (ot * 16 + lo) * 64;

  v8f acc = {};
#pragma unroll
  for (int kb = 0; kb < 64; kb += 32) {
    v16bf a, b;
#pragma unroll
    for (int i = 0; i < 8; ++i) {
      bf16x2 ap = ld2(arow + kb + ak_of(i, hi));  // w2 zero for k>=48
      a[2 * i] = ap.x; a[2 * i + 1] = ap.y;
      int kB = kb + bk_of(i, hi);
      if (kb < 32) {  // compile-time: k <= 30 < 48, plain contiguous loads
        f32x2 sp = *(const f32x2*)(srow + kB);
        b[2 * i] = (bf16)sp.x; b[2 * i + 1] = (bf16)sp.y;
      } else {        // last block: clamp address, select value to zero
        int kc = kB < 46 ? kB : 46;
        f32x2 sp = *(const f32x2*)(srow + kc);
        b[2 * i]     = (kB < 48)     ? (bf16)sp.x : (bf16)0.f;
        b[2 * i + 1] = (kB + 1 < 48) ? (bf16)sp.y : (bf16)0.f;
      }
    }
    acc = WMMA_BF16(a, b, acc);
  }
#pragma unroll
  for (int r = 0; r < 8; ++r) {
    int o = ot * 16 + r + 8 * hi;
    float sc = gamma[o] * rsqrtf(var[o] + 1e-5f);
    long xi = (((long)n * CO + o) * T_DIM + t) * V_DIM + v;
    float val = (acc[r] + outb[o] - mean[o]) * sc + beta[o] + x[xi];
    y[xi] = fmaxf(val, 0.f);
  }
}

// ---------------------------------------------------------------------------
// Workspace layout (bytes). s2 overlays xb/a1t (dead before GEMM2 writes s2).
// Total ~256 MB.
// ---------------------------------------------------------------------------
static constexpr size_t alignup(size_t x) { return (x + 255) & ~(size_t)255; }
static constexpr size_t XB_BYTES   = (size_t)NB * TV * C_IN * 2;          //  52.4 MB
static constexpr size_t A1T_BYTES  = (size_t)NB * 80 * KCT * 2;           //  41.9 MB
static constexpr size_t S2_BYTES   = (size_t)NB * T_DIM * V_DIM * 48 * 4; //  78.6 MB
static constexpr size_t P2_BYTES   = (size_t)NB * 64 * T_DIM * PW * 2;    // 159.4 MB
static constexpr size_t LG_BYTES   = (size_t)NB * W3 * W3 * 4;
static constexpr size_t ATTT_BYTES = (size_t)NB * 80 * AKP * 2;
static constexpr size_t AEFF_BYTES = (size_t)3 * 80 * AKP * 2;
static constexpr size_t WCAT_BYTES = (size_t)80 * C_IN * 2;
static constexpr size_t W2_BYTES   = (size_t)CO * 64 * 2;
static constexpr size_t BC_BYTES   = (size_t)80 * 4;

static constexpr size_t OFF_XB   = 0;
static constexpr size_t OFF_A1T  = alignup(OFF_XB + XB_BYTES);
static constexpr size_t OFF_S2   = 0;  // alias: s2 written after xb/a1t are dead
static constexpr size_t REGION_A = alignup(OFF_A1T + A1T_BYTES) > alignup(S2_BYTES)
                                       ? alignup(OFF_A1T + A1T_BYTES)
                                       : alignup(S2_BYTES);
static constexpr size_t OFF_P2   = REGION_A;
static constexpr size_t OFF_LG   = alignup(OFF_P2 + P2_BYTES);
static constexpr size_t OFF_ATTT = alignup(OFF_LG + LG_BYTES);
static constexpr size_t OFF_AEFF = alignup(OFF_ATTT + ATTT_BYTES);
static constexpr size_t OFF_WCAT = alignup(OFF_AEFF + AEFF_BYTES);
static constexpr size_t OFF_W2   = alignup(OFF_WCAT + WCAT_BYTES);
static constexpr size_t OFF_BC   = alignup(OFF_W2 + W2_BYTES);

extern "C" void kernel_launch(void* const* d_in, const int* in_sizes, int n_in,
                              void* d_out, int out_size, void* d_ws, size_t ws_size,
                              hipStream_t stream) {
  const float* x     = (const float*)d_in[0];
  const float* A     = (const float*)d_in[1];
  const float* PA    = (const float*)d_in[2];
  const float* wa    = (const float*)d_in[3];
  const float* ab    = (const float*)d_in[4];
  const float* wd    = (const float*)d_in[5];
  const float* db    = (const float*)d_in[6];
  const float* wc    = (const float*)d_in[7];
  const float* cb    = (const float*)d_in[8];
  const float* ow    = (const float*)d_in[9];
  const float* outb  = (const float*)d_in[10];
  const float* gamma = (const float*)d_in[11];
  const float* beta  = (const float*)d_in[12];
  const float* mean  = (const float*)d_in[13];
  const float* var   = (const float*)d_in[14];
  float* y = (float*)d_out;

  char* ws = (char*)d_ws;
  bf16*  xb     = (bf16*)(ws + OFF_XB);
  bf16*  a1t    = (bf16*)(ws + OFF_A1T);
  float* s2     = (float*)(ws + OFF_S2);
  bf16*  P2     = (bf16*)(ws + OFF_P2);
  float* logits = (float*)(ws + OFF_LG);
  bf16*  attT   = (bf16*)(ws + OFF_ATTT);
  bf16*  aeffT  = (bf16*)(ws + OFF_AEFF);
  bf16*  wcat   = (bf16*)(ws + OFF_WCAT);
  bf16*  w2     = (bf16*)(ws + OFF_W2);
  float* bcat   = (float*)(ws + OFF_BC);

  k_prep<<<512, 256, 0, stream>>>(A, PA, wa, wd, wc, ow, ab, db, cb,
                                  aeffT, wcat, w2, attT, bcat);
  k_xpose<<<(int)(((long)NB * TV * C_IN + 255) / 256), 256, 0, stream>>>(x, xb);
  // 32 * 5 * 2400 tiles, 8 waves/block
  k_gemm1<<<48000, 256, 0, stream>>>(xb, wcat, bcat, P2, a1t);
  // 32 * 25 tiles
  k_att<<<100, 256, 0, stream>>>(a1t, logits);
  // one block per (n, column j)
  k_softmax<<<NB * W3, 128, 0, stream>>>(logits, attT);
  // 32 * 512 * 5 tiles
  k_gemm2<<<10240, 256, 0, stream>>>(P2, attT, aeffT, s2);
  // 32 * 4 * 800 tiles
  k_out<<<12800, 256, 0, stream>>>(s2, w2, outb, gamma, beta, mean, var, x, y);
}
